// SAGE_721554505786
// MI455X (gfx1250) — matrix operand order repaired and checked by reading, not compiled, not run on hardware
//
#include <hip/hip_runtime.h>

// ---------------------------------------------------------------------------
// MI455X / gfx1250 implementation of a 2-layer LSTM-SAGE GNN.
// - All heavy GEMMs: v_wmma_f32_16x16x32_bf16 (wave32 WMMA), weight B-frags
//   re-loaded from global/L2 at each of the 16 recurrence steps.
// - Gate nonlinearities: gfx1250 hardware TRANS ops (v_tanh_f32).
// - Layer-2 neighbor gather: bf16 h1 + async global->LDS copies (ASYNCcnt).
// ---------------------------------------------------------------------------

typedef __attribute__((ext_vector_type(16))) __bf16 v16bf;
typedef __attribute__((ext_vector_type(8)))  __bf16 v8bf;
typedef __attribute__((ext_vector_type(8)))  float  v8f;
typedef int v2i __attribute__((vector_size(8)));   // b64 payload type for async LDS loads

static constexpr int N_NODES = 32768;
static constexpr int KNBR    = 16;
static constexpr int HDIM    = 128;
static constexpr int NGRAPH  = 1024;
static constexpr int TDIM    = 2;

static constexpr float LOG2E = 1.44269504088896340736f;

#if __has_builtin(__builtin_amdgcn_global_load_async_to_lds_b64) && \
    __has_builtin(__builtin_amdgcn_s_wait_asynccnt)
#define USE_ASYNC_GATHER 1
typedef __attribute__((address_space(1))) v2i* gv2i_p;   // global (AS1)
typedef __attribute__((address_space(3))) v2i* lv2i_p;   // LDS    (AS3)
#else
#define USE_ASYNC_GATHER 0
#endif

__device__ __forceinline__ float fast_exp2(float x) { return __builtin_amdgcn_exp2f(x); }
__device__ __forceinline__ float fast_rcp(float x)  { return __builtin_amdgcn_rcpf(x); }

__device__ __forceinline__ float fast_tanh(float x) {
#if __has_builtin(__builtin_amdgcn_tanhf)
    return __builtin_amdgcn_tanhf(x);          // gfx1250 V_TANH_F32 (TRANS)
#else
    const float e = fast_exp2((2.0f * LOG2E) * x);
    return 1.0f - 2.0f * fast_rcp(e + 1.0f);
#endif
}

__device__ __forceinline__ float sigm(float x) {
#if __has_builtin(__builtin_amdgcn_tanhf)
    return __builtin_amdgcn_tanhf(0.5f * x) * 0.5f + 0.5f;
#else
    return fast_rcp(1.0f + fast_exp2(-LOG2E * x));
#endif
}

__device__ __forceinline__ v8f wmma_bf16(v16bf a, v16bf b, v8f c) {
    // D = A(16x32 bf16) * B(32x16 bf16) + C(16x16 f32)
    return __builtin_amdgcn_wmma_f32_16x16x32_bf16(
        /*neg_a=*/false, a, /*neg_b=*/false, b,
        /*c_mod=*/(short)0, c, /*reuse_a=*/false, /*reuse_b=*/false);
}

// A fragment (16x32 bf16 tile, row-major with row stride FIN, from LDS).
// Lane l (l<16): M=l, K = {0..15}; lane l+16: M=l, K={16..31} (matches B frag).
template <int FIN>
__device__ __forceinline__ v16bf load_a_frag(const __bf16* __restrict__ base,
                                             int l15, int half, int kk) {
    const __bf16* rp = base + l15 * FIN + kk * 32 + half * 16;
    v8bf lo = *(const v8bf*)rp;          // K = kk*32 + half*16 .. +7
    v8bf hi = *(const v8bf*)(rp + 8);    // K = kk*32 + half*16 + 8 .. +15
    v16bf a;
#pragma unroll
    for (int i = 0; i < 8; ++i) { a[i] = lo[i]; a[i + 8] = hi[i]; }
    return a;
}

// B fragment from a transposed weight matrix WT[cols][FIN] (bf16, global/L2).
template <int FIN>
__device__ __forceinline__ v16bf load_b_frag(const __bf16* __restrict__ WT,
                                             int colbase, int l15, int half, int kk) {
    return *(const v16bf*)(WT + (size_t)(colbase + l15) * FIN + kk * 32 + half * 16);
}

// ---------------------------------------------------------------------------
// Weight convert+transpose: src f32 [R][C] -> dst bf16 [C][R]
// ---------------------------------------------------------------------------
__global__ void convert_transpose_bf16(const float* __restrict__ src,
                                       __bf16* __restrict__ dst, int R, int C) {
    int i = blockIdx.x * blockDim.x + threadIdx.x;
    if (i < R * C) {
        int r = i / C, c = i % C;
        dst[(size_t)c * R + r] = (__bf16)src[i];
    }
}

// ---------------------------------------------------------------------------
// Fused SAGE layer. XT (input elem) = float (layer 1) or __bf16 (layer 2);
// OT (output elem) = __bf16 (h1) or float (h2). Out dim is always 128.
// Per block: ROWS = 2048/FIN nodes, 256 threads = 8 waves (wave32).
// ---------------------------------------------------------------------------
template <int FIN, typename XT, typename OT>
__global__ __launch_bounds__(256) void sage_lstm_layer(
    const XT*     __restrict__ x,      // [N, FIN] node features
    const int*    __restrict__ nbr,    // [N, 16]
    const __bf16* __restrict__ WihT,   // [4*FIN][FIN]  (transposed, bf16)
    const __bf16* __restrict__ WhhT,   // [4*FIN][FIN]
    const float*  __restrict__ blstm,  // [4*FIN]
    const __bf16* __restrict__ WsT,    // [128][FIN]
    const float*  __restrict__ bs,     // [128]
    const __bf16* __restrict__ WnT,    // [128][FIN]
    const float*  __restrict__ bn,     // [128]
    OT*           __restrict__ out)    // [N, 128]
{
    constexpr int ROWS = 2048 / FIN;   // nodes per workgroup (32 or 16)
    constexpr int WPR  = FIN / 16;     // waves per row tile (4 or 8)
    constexpr int RT   = 8 / WPR;      // row tiles per workgroup (2 or 1)
    constexpr int KT   = FIN / 32;     // WMMA k-steps (2 or 4)
    constexpr int VEC  = FIN / 32;     // elems per lane per gathered row

    extern __shared__ __bf16 smem[];
    __bf16* xseq  = smem;                          // [16][ROWS][FIN]  neighbor seq
    __bf16* xself = xseq + KNBR * ROWS * FIN;      // [ROWS][FIN]      self features
    __bf16* hbuf  = xself + ROWS * FIN;            // [ROWS][FIN]      LSTM hidden

    const int tid  = threadIdx.x;
    const int wave = tid >> 5;
    const int lane = tid & 31;
    const int half = lane >> 4;
    const int l15  = lane & 15;
    const int nodeBase = blockIdx.x * ROWS;

    const int rt   = wave / WPR;                // row tile of this wave
    const int unit = wave % WPR;                // hidden-unit column tile
    const int col  = unit * 16 + l15;           // hidden unit column for this lane

    // Warm L2/WGP$ for this wave's weight fragments (global_prefetch_b8).
#pragma unroll
    for (int g2 = 0; g2 < 4; ++g2) {
        __builtin_prefetch(WihT + (size_t)(g2 * FIN + unit * 16 + l15) * FIN, 0, 0);
        __builtin_prefetch(WhhT + (size_t)(g2 * FIN + unit * 16 + l15) * FIN, 0, 0);
    }
    __builtin_prefetch(WsT + (size_t)(wave * 16 + l15) * FIN, 0, 0);
    __builtin_prefetch(WnT + (size_t)(wave * 16 + l15) * FIN, 0, 0);

    // ---- gather neighbor sequences + self rows into LDS ----
    for (int r = wave; r < ROWS * KNBR; r += 8) {
        const int nl = r >> 4;          // local node
        const int t  = r & 15;          // neighbor slot (= LSTM time step)
        const int idx = nbr[(size_t)(nodeBase + nl) * KNBR + t];
        const XT* src = x + (size_t)idx * FIN + lane * VEC;
        __bf16* dst = xseq + ((size_t)t * ROWS + nl) * FIN + lane * VEC;
        if constexpr (sizeof(XT) == 4) {            // f32 input: convert to bf16
            if constexpr (VEC == 4) {
                const float4 v = *(const float4*)src;
                dst[0] = (__bf16)v.x; dst[1] = (__bf16)v.y;
                dst[2] = (__bf16)v.z; dst[3] = (__bf16)v.w;
            } else {
                const float2 v = *(const float2*)src;
                dst[0] = (__bf16)v.x; dst[1] = (__bf16)v.y;
            }
        } else {                                    // bf16 input: raw copy
#if USE_ASYNC_GATHER
            if constexpr (VEC == 4) {               // 8B/lane -> async b64 to LDS
                __builtin_amdgcn_global_load_async_to_lds_b64(
                    (gv2i_p)src, (lv2i_p)dst, 0, 0);
            } else {
                *(uint*)dst = *(const uint*)src;
            }
#else
            if constexpr (VEC == 4) *(uint2*)dst = *(const uint2*)src;
            else                    *(uint*)dst  = *(const uint*)src;
#endif
        }
    }
    for (int r = wave; r < ROWS; r += 8) {
        const XT* src = x + (size_t)(nodeBase + r) * FIN + lane * VEC;
        __bf16* dst = xself + (size_t)r * FIN + lane * VEC;
        if constexpr (sizeof(XT) == 4) {
            if constexpr (VEC == 4) {
                const float4 v = *(const float4*)src;
                dst[0] = (__bf16)v.x; dst[1] = (__bf16)v.y;
                dst[2] = (__bf16)v.z; dst[3] = (__bf16)v.w;
            } else {
                const float2 v = *(const float2*)src;
                dst[0] = (__bf16)v.x; dst[1] = (__bf16)v.y;
            }
        } else {
#if USE_ASYNC_GATHER
            if constexpr (VEC == 4) {
                __builtin_amdgcn_global_load_async_to_lds_b64(
                    (gv2i_p)src, (lv2i_p)dst, 0, 0);
            } else {
                *(uint*)dst = *(const uint*)src;
            }
#else
            if constexpr (VEC == 4) *(uint2*)dst = *(const uint2*)src;
            else                    *(uint*)dst  = *(const uint*)src;
#endif
        }
    }
    // ---- h0 = 0 ----
    for (int i = tid; i < ROWS * FIN; i += 256) hbuf[i] = (__bf16)0.0f;
#if USE_ASYNC_GATHER
    if constexpr (sizeof(XT) == 2) __builtin_amdgcn_s_wait_asynccnt(0);
#endif
    __syncthreads();

    const float bi  = blstm[0 * FIN + col];
    const float bf_ = blstm[1 * FIN + col];
    const float bg  = blstm[2 * FIN + col];
    const float bo  = blstm[3 * FIN + col];

    float cst[8];                                // cell state lives in registers
#pragma unroll
    for (int j = 0; j < 8; ++j) cst[j] = 0.0f;

    // ---- 16-step LSTM recurrence over the neighbor sequence ----
    for (int t = 0; t < KNBR; ++t) {
        v8f acc[4];
#pragma unroll
        for (int g = 0; g < 4; ++g)
#pragma unroll
            for (int j = 0; j < 8; ++j) acc[g][j] = 0.0f;

        const __bf16* xt = xseq + (size_t)t * ROWS * FIN + rt * 16 * FIN;
        const __bf16* hp = hbuf + rt * 16 * FIN;
#pragma unroll
        for (int kk = 0; kk < KT; ++kk) {
            const v16bf ax = load_a_frag<FIN>(xt, l15, half, kk);
            const v16bf ah = load_a_frag<FIN>(hp, l15, half, kk);
#pragma unroll
            for (int g = 0; g < 4; ++g) {
                const v16bf bx = load_b_frag<FIN>(WihT, g * FIN + unit * 16, l15, half, kk);
                acc[g] = wmma_bf16(ax, bx, acc[g]);
                const v16bf bh = load_b_frag<FIN>(WhhT, g * FIN + unit * 16, l15, half, kk);
                acc[g] = wmma_bf16(ah, bh, acc[g]);
            }
        }
        __syncthreads();   // everyone done reading hbuf before it is rewritten

        // torch gate order i,f,g,o ; c = sig(f)*c + sig(i)*tanh(g); h = sig(o)*tanh(c)
#pragma unroll
        for (int j = 0; j < 8; ++j) {
            const float iv = sigm(acc[0][j] + bi);
            const float fv = sigm(acc[1][j] + bf_);
            const float gv = fast_tanh(acc[2][j] + bg);
            const float ov = sigm(acc[3][j] + bo);
            const float c  = fv * cst[j] + iv * gv;
            cst[j] = c;
            const float hv = ov * fast_tanh(c);
            const int m = j + half * 8;          // C-tile row (VGPR j -> M=j / j+8)
            hbuf[(rt * 16 + m) * FIN + col] = (__bf16)hv;
        }
        __syncthreads();
    }

    // ---- out = sigmoid(x_self @ Ws + bs + h_lstm @ Wn + bn), 128 cols ----
    const int ocol = wave * 16 + l15;            // each wave owns one 16-col tile
    const float bso = bs[ocol] + bn[ocol];
    for (int rt2 = 0; rt2 < RT; ++rt2) {
        v8f acc;
#pragma unroll
        for (int j = 0; j < 8; ++j) acc[j] = 0.0f;
        const __bf16* xs = xself + rt2 * 16 * FIN;
        const __bf16* hp = hbuf + rt2 * 16 * FIN;
#pragma unroll
        for (int kk = 0; kk < KT; ++kk) {
            acc = wmma_bf16(load_a_frag<FIN>(xs, l15, half, kk),
                            load_b_frag<FIN>(WsT, wave * 16, l15, half, kk), acc);
            acc = wmma_bf16(load_a_frag<FIN>(hp, l15, half, kk),
                            load_b_frag<FIN>(WnT, wave * 16, l15, half, kk), acc);
        }
#pragma unroll
        for (int j = 0; j < 8; ++j) {
            const int m = j + half * 8;
            out[(size_t)(nodeBase + rt2 * 16 + m) * HDIM + ocol] = (OT)sigm(acc[j] + bso);
        }
    }
}

// ---------------------------------------------------------------------------
// WeightedSumAndMax readout. gid = (arange(N)*NG)//N is contiguous with exactly
// N/NG = 32 nodes per graph -> one block per graph, no atomics needed.
// ---------------------------------------------------------------------------
__global__ __launch_bounds__(128) void readout_kernel(
    const float* __restrict__ h2,     // [N,128]
    const float* __restrict__ rwW,    // [128]
    const float* __restrict__ rwb,    // [1]
    float*       __restrict__ gemb)   // [G,256]
{
    __shared__ float part[128];
    __shared__ float wld[32];
    const int g = blockIdx.x;
    const int t = threadIdx.x;
    const int nbase = g * 32;

    {
        const int nl = t >> 2, q = t & 3;
        const float* hrow = h2 + (size_t)(nbase + nl) * HDIM + q * 32;
        float s = 0.0f;
#pragma unroll
        for (int c = 0; c < 32; ++c) s += hrow[c] * rwW[q * 32 + c];
        part[t] = s;
    }
    __syncthreads();
    if (t < 32) {
        const float s = part[t * 4] + part[t * 4 + 1] + part[t * 4 + 2] + part[t * 4 + 3] + rwb[0];
        wld[t] = sigm(s);
    }
    __syncthreads();

    float ws = 0.0f, mx = -3.402823466e38f;
    for (int n = 0; n < 32; ++n) {
        const float v = h2[(size_t)(nbase + n) * HDIM + t];
        ws += wld[n] * v;
        mx = fmaxf(mx, v);
    }
    gemb[(size_t)g * 256 + t]       = ws;
    gemb[(size_t)g * 256 + 128 + t] = mx;
}

// ---------------------------------------------------------------------------
// Head MLP: y = sigmoid(sigmoid(gemb @ h1_W + h1_b) @ h2_W + h2_b)  (tiny)
// ---------------------------------------------------------------------------
__global__ __launch_bounds__(128) void head_kernel(
    const float* __restrict__ gemb,   // [G,256]
    const float* __restrict__ h1W,    // [256][128]
    const float* __restrict__ h1b,    // [128]
    const float* __restrict__ h2W,    // [128][2]
    const float* __restrict__ h2b,    // [2]
    float*       __restrict__ y)      // [G,2]
{
    __shared__ float ge[256];
    __shared__ float y1[128];
    const int g = blockIdx.x;
    const int t = threadIdx.x;
    ge[t]       = gemb[(size_t)g * 256 + t];
    ge[128 + t] = gemb[(size_t)g * 256 + 128 + t];
    __syncthreads();
    float s = h1b[t];
    for (int k = 0; k < 256; ++k) s += ge[k] * h1W[(size_t)k * HDIM + t];
    y1[t] = sigm(s);
    __syncthreads();
    if (t < TDIM) {
        float s2 = h2b[t];
        for (int k = 0; k < 128; ++k) s2 += y1[k] * h2W[(size_t)k * TDIM + t];
        y[(size_t)g * TDIM + t] = sigm(s2);
    }
}

// ---------------------------------------------------------------------------
// Workspace layout (bytes, 256-aligned). h1 is stored bf16 (feeds only WMMA).
// ---------------------------------------------------------------------------
static constexpr size_t OFF_WIH1 = 0;                         // 256*64  bf16
static constexpr size_t OFF_WHH1 = 32768;                     // 256*64  bf16
static constexpr size_t OFF_WS1  = 65536;                     // 128*64  bf16
static constexpr size_t OFF_WN1  = 81920;                     // 128*64  bf16
static constexpr size_t OFF_WIH2 = 98304;                     // 512*128 bf16
static constexpr size_t OFF_WHH2 = 229376;                    // 512*128 bf16
static constexpr size_t OFF_WS2  = 360448;                    // 128*128 bf16
static constexpr size_t OFF_WN2  = 393216;                    // 128*128 bf16
static constexpr size_t OFF_H1   = 425984;                    // N*128 bf16
static constexpr size_t OFF_H2   = OFF_H1 + (size_t)N_NODES * HDIM * 2;
static constexpr size_t OFF_GEMB = OFF_H2 + (size_t)N_NODES * HDIM * 4;

extern "C" void kernel_launch(void* const* d_in, const int* in_sizes, int n_in,
                              void* d_out, int out_size, void* d_ws, size_t ws_size,
                              hipStream_t stream) {
    (void)in_sizes; (void)n_in; (void)out_size; (void)ws_size;

    const float* n_feat = (const float*)d_in[0];
    const int*   nbr    = (const int*)d_in[1];
    // d_in[2]=gid (contiguous by construction), d_in[3]=G (=1024) : not needed
    const float* l1Wih  = (const float*)d_in[4];
    const float* l1Whh  = (const float*)d_in[5];
    const float* l1b    = (const float*)d_in[6];
    const float* fcs1W  = (const float*)d_in[7];
    const float* fcs1b  = (const float*)d_in[8];
    const float* fcn1W  = (const float*)d_in[9];
    const float* fcn1b  = (const float*)d_in[10];
    const float* l2Wih  = (const float*)d_in[11];
    const float* l2Whh  = (const float*)d_in[12];
    const float* l2b    = (const float*)d_in[13];
    const float* fcs2W  = (const float*)d_in[14];
    const float* fcs2b  = (const float*)d_in[15];
    const float* fcn2W  = (const float*)d_in[16];
    const float* fcn2b  = (const float*)d_in[17];
    const float* rwW    = (const float*)d_in[18];
    const float* rwb    = (const float*)d_in[19];
    const float* h1W    = (const float*)d_in[20];
    const float* h1b    = (const float*)d_in[21];
    const float* h2W    = (const float*)d_in[22];
    const float* h2b    = (const float*)d_in[23];

    char* ws = (char*)d_ws;
    __bf16* WihT1 = (__bf16*)(ws + OFF_WIH1);
    __bf16* WhhT1 = (__bf16*)(ws + OFF_WHH1);
    __bf16* WsT1  = (__bf16*)(ws + OFF_WS1);
    __bf16* WnT1  = (__bf16*)(ws + OFF_WN1);
    __bf16* WihT2 = (__bf16*)(ws + OFF_WIH2);
    __bf16* WhhT2 = (__bf16*)(ws + OFF_WHH2);
    __bf16* WsT2  = (__bf16*)(ws + OFF_WS2);
    __bf16* WnT2  = (__bf16*)(ws + OFF_WN2);
    __bf16* h1    = (__bf16*)(ws + OFF_H1);
    float*  h2    = (float*)(ws + OFF_H2);
    float*  gemb  = (float*)(ws + OFF_GEMB);

    // 1) weights -> bf16 transposed (L2-resident, ~0.5 MB total)
    convert_transpose_bf16<<<(64 * 256 + 255) / 256, 256, 0, stream>>>(l1Wih, WihT1, 64, 256);
    convert_transpose_bf16<<<(64 * 256 + 255) / 256, 256, 0, stream>>>(l1Whh, WhhT1, 64, 256);
    convert_transpose_bf16<<<(64 * 128 + 255) / 256, 256, 0, stream>>>(fcs1W, WsT1, 64, 128);
    convert_transpose_bf16<<<(64 * 128 + 255) / 256, 256, 0, stream>>>(fcn1W, WnT1, 64, 128);
    convert_transpose_bf16<<<(128 * 512 + 255) / 256, 256, 0, stream>>>(l2Wih, WihT2, 128, 512);
    convert_transpose_bf16<<<(128 * 512 + 255) / 256, 256, 0, stream>>>(l2Whh, WhhT2, 128, 512);
    convert_transpose_bf16<<<(128 * 128 + 255) / 256, 256, 0, stream>>>(fcs2W, WsT2, 128, 128);
    convert_transpose_bf16<<<(128 * 128 + 255) / 256, 256, 0, stream>>>(fcn2W, WnT2, 128, 128);

    // 2) fused SAGE layers (xseq 64KB + xself 4KB + h 4KB of dynamic LDS)
    const size_t shmem = (size_t)(KNBR + 2) * 2048 * sizeof(__bf16);   // 73728 B
    sage_lstm_layer<64, float, __bf16><<<N_NODES / 32, 256, shmem, stream>>>(
        n_feat, nbr, WihT1, WhhT1, l1b, WsT1, fcs1b, WnT1, fcn1b, h1);
    sage_lstm_layer<128, __bf16, float><<<N_NODES / 16, 256, shmem, stream>>>(
        h1, nbr, WihT2, WhhT2, l2b, WsT2, fcs2b, WnT2, fcn2b, h2);

    // 3) readout + head
    readout_kernel<<<NGRAPH, 128, 0, stream>>>(h2, rwW, rwb, gemb);
    head_kernel<<<NGRAPH, 128, 0, stream>>>(gemb, h1W, h1b, h2W, h2b, (float*)d_out);
}
